// MHSA_10514079941074
// MI455X (gfx1250) — matrix operand order repaired and checked by reading, not compiled, not run on hardware
//
#include <hip/hip_runtime.h>
#include <hip/hip_bf16.h>

typedef _Float16 v16h __attribute__((ext_vector_type(16)));
typedef _Float16 v8h  __attribute__((ext_vector_type(8)));
typedef float    v8f  __attribute__((ext_vector_type(8)));

#define EMBD  512
#define HEADS 8
#define HEAD  64
#define BATCH 4
#define SEQ   2048
#define ROWS  (BATCH * SEQ)   // 8192

static __device__ __forceinline__ v8f wmma16(v16h a, v16h b, v8f c) {
  // v_wmma_f32_16x16x32_f16  (wave32, D = A*B + C, f32 accum)
  return __builtin_amdgcn_wmma_f32_16x16x32_f16(false, a, false, b, (short)0, c,
                                                false, false);
}

// A fragment 16x32 f16, row-major source. `p` points at (row m = lane&15, col k0).
// Lanes 0-15 hold K=0..15, lanes 16-31 hold K=16..31.
static __device__ __forceinline__ v16h load_a_frag(const _Float16* p, int hsel) {
  v8h lo = *(const v8h*)(p + hsel * 16);
  v8h hi = *(const v8h*)(p + hsel * 16 + 8);
  v16h a;
#pragma unroll
  for (int i = 0; i < 8; ++i) { a[i] = lo[i]; a[8 + i] = hi[i]; }
  return a;
}

//==============================================================
// 0) convert the four 512x512 f32 weight matrices to f16
//==============================================================
__global__ void __launch_bounds__(256) cvt_w_kernel(
    const float* __restrict__ wq, const float* __restrict__ wk,
    const float* __restrict__ wv, const float* __restrict__ wo,
    _Float16* __restrict__ wqh, _Float16* __restrict__ wkh,
    _Float16* __restrict__ wvh, _Float16* __restrict__ woh) {
  int i = blockIdx.x * 256 + threadIdx.x;
  if (i < EMBD * EMBD) {
    wqh[i] = (_Float16)wq[i];
    wkh[i] = (_Float16)wk[i];
    wvh[i] = (_Float16)wv[i];
    woh[i] = (_Float16)wo[i];
  }
}

//==============================================================
// 1) LayerNorm, f32 in -> f16 out (one block per row)
//==============================================================
__global__ void __launch_bounds__(256) ln_kernel(
    const float* __restrict__ x, const float* __restrict__ gamma,
    const float* __restrict__ beta, _Float16* __restrict__ normh) {
  __shared__ float red[256];
  int row = blockIdx.x;
  int t = threadIdx.x;
  const float* xr = x + (size_t)row * EMBD;
  float v0 = xr[t], v1 = xr[t + 256];
  red[t] = v0 + v1;
  __syncthreads();
  for (int s = 128; s > 0; s >>= 1) {
    if (t < s) red[t] += red[t + s];
    __syncthreads();
  }
  float mu = red[0] * (1.0f / EMBD);
  __syncthreads();
  float d0 = v0 - mu, d1 = v1 - mu;
  red[t] = d0 * d0 + d1 * d1;
  __syncthreads();
  for (int s = 128; s > 0; s >>= 1) {
    if (t < s) red[t] += red[t + s];
    __syncthreads();
  }
  float rstd = rsqrtf(red[0] * (1.0f / EMBD) + 1e-5f);
  _Float16* nr = normh + (size_t)row * EMBD;
  nr[t]       = (_Float16)(d0 * rstd * gamma[t] + beta[t]);
  nr[t + 256] = (_Float16)(d1 * rstd * gamma[t + 256] + beta[t + 256]);
}

//==============================================================
// 2) fused Q/K/V projection: C[m,e] = sum_k norm[m,k]*W[e,k] + b[e]
//    Each wave computes a 16x64 output tile, software-pipelined AND
//    fully unrolled so the double-buffer rotation is renamed away.
//    V is stored transposed as Vt[b,h,d,m] for contiguous P*V frags.
//==============================================================
__global__ void __launch_bounds__(256) proj_qkv_kernel(
    const _Float16* __restrict__ normh,
    const _Float16* __restrict__ wqh, const _Float16* __restrict__ wkh,
    const _Float16* __restrict__ wvh,
    const float* __restrict__ bq, const float* __restrict__ bk,
    const float* __restrict__ bv,
    _Float16* __restrict__ Qh, _Float16* __restrict__ Kh,
    _Float16* __restrict__ Vt) {
  int t = threadIdx.x;
  int l = t & 31;
  int w = blockIdx.x * 8 + (t >> 5);   // 12288 waves total
  int target = w >> 12;                // 0=Q 1=K 2=V
  int rem = w & 4095;
  int mt = rem >> 3;                   // 512 m-tiles
  int es = rem & 7;                    // 8 e-supertiles of 64
  int m0 = mt * 16, e0 = es * 64;
  int mrow = l & 15, hsel = l >> 4, ncol = l & 15, rsel = l >> 4;

  const _Float16* W = (target == 0) ? wqh : (target == 1) ? wkh : wvh;
  const float* bias = (target == 0) ? bq : (target == 1) ? bk : bv;
  const _Float16* arow = normh + (size_t)(m0 + mrow) * EMBD;
  const _Float16* wrow = W + (size_t)(e0 + ncol) * EMBD + hsel * 16;

  v8f acc[4];
#pragma unroll
  for (int n = 0; n < 4; ++n)
#pragma unroll
    for (int j = 0; j < 8; ++j) acc[n][j] = 0.0f;

  // prologue: stage k-step 0
  v16h a_cur = load_a_frag(arow, hsel);
  v16h b_cur[4];
#pragma unroll
  for (int n = 0; n < 4; ++n)
    b_cur[n] = *(const v16h*)(wrow + (size_t)n * 16 * EMBD);

#pragma unroll
  for (int ks = 0; ks < EMBD / 32; ++ks) {
    v16h a_nxt = a_cur;
    v16h b_nxt[4];
#pragma unroll
    for (int n = 0; n < 4; ++n) b_nxt[n] = b_cur[n];
    if (ks < EMBD / 32 - 1) {
      int k1 = (ks + 1) * 32;
      a_nxt = load_a_frag(arow + k1, hsel);
#pragma unroll
      for (int n = 0; n < 4; ++n)
        b_nxt[n] = *(const v16h*)(wrow + (size_t)n * 16 * EMBD + k1);
    }
#pragma unroll
    for (int n = 0; n < 4; ++n) acc[n] = wmma16(a_cur, b_cur[n], acc[n]);
    a_cur = a_nxt;
#pragma unroll
    for (int n = 0; n < 4; ++n) b_cur[n] = b_nxt[n];
  }

#pragma unroll
  for (int n = 0; n < 4; ++n) {
    int ge = e0 + n * 16 + ncol;
    float bb = bias[ge];
    if (target < 2) {
      _Float16* dst = (target == 0) ? Qh : Kh;
#pragma unroll
      for (int j = 0; j < 8; ++j) {
        int gm = m0 + rsel * 8 + j;
        dst[(size_t)gm * EMBD + ge] = (_Float16)(acc[n][j] + bb);
      }
    } else {
      int bb_idx = m0 >> 11;                 // batch
      int mloc = (m0 & 2047) + rsel * 8;     // row inside sequence
      int h = ge >> 6, d = ge & 63;
      v8h pk;
#pragma unroll
      for (int j = 0; j < 8; ++j) pk[j] = (_Float16)(acc[n][j] + bb);
      *(v8h*)(Vt + (((size_t)(bb_idx * HEADS + h) * HEAD + d) * SEQ + mloc)) = pk;
    }
  }
}

//==============================================================
// 3) attention: one workgroup = 16 query rows of one (b,h).
//    Scores stay LDS-resident (16 x 2048 f32 = 128 KB of the
//    320 KB WGP LDS). Exact softmax, diag masked. attn written
//    once, coalesced. P*V via WMMA with per-row 1/sum post-scale.
//==============================================================
#define SMEM_S      (16 * 2048)            // scores / probs
#define SMEM_RED    256                    // 16x16 partials
#define SMEM_STAT   32                     // [0..15]=inv_sum [16..31]=rowmax
#define SMEM_PVRED  (4 * 32 * 8)           // P*V cross-wave reduce
#define SMEM_FLOATS (SMEM_S + SMEM_RED + SMEM_STAT + SMEM_PVRED)

__global__ void __launch_bounds__(256) attn_kernel(
    const _Float16* __restrict__ Qh, const _Float16* __restrict__ Kh,
    const _Float16* __restrict__ Vt,
    float* __restrict__ attn, _Float16* __restrict__ ctxh) {
  extern __shared__ float smem[];
  float* S       = smem;
  float* red     = smem + SMEM_S;
  float* rowstat = red + SMEM_RED;
  float* pvred   = rowstat + SMEM_STAT;

  int t = threadIdx.x, l = t & 31, wv = t >> 5;
  int bid = blockIdx.x;
  int mt = bid & 127, h = (bid >> 7) & 7, b = bid >> 10;
  int m0 = mt * 16;
  int mrow = l & 15, hsel = l >> 4, ncol = l & 15, rsel = l >> 4;
  const float NEG_INF = -__builtin_inff();

  //---- phase 1: S = Q*K^T / 8, diagonal masked, 1-deep pipeline -
  const _Float16* qrow = Qh + (size_t)(b * SEQ + m0 + mrow) * EMBD + h * HEAD;
  v16h aq0 = load_a_frag(qrow, hsel);
  v16h aq1 = load_a_frag(qrow + 32, hsel);
  const _Float16* kbase = Kh + (size_t)(b * SEQ) * EMBD + h * HEAD + hsel * 16;

  const _Float16* kr = kbase + (size_t)(wv * 16 + ncol) * EMBD;
  v16h bk0 = *(const v16h*)(kr);
  v16h bk1 = *(const v16h*)(kr + 32);
#pragma unroll
  for (int it = 0; it < SEQ / 16 / 8; ++it) {
    int n0 = (wv + it * 8) * 16;
    v16h nb0 = bk0, nb1 = bk1;
    if (it < SEQ / 16 / 8 - 1) {
      const _Float16* krn = kbase + (size_t)(n0 + 128 + ncol) * EMBD;
      nb0 = *(const v16h*)(krn);
      nb1 = *(const v16h*)(krn + 32);
      // stream hint two tiles ahead (global_prefetch_b8)
      __builtin_prefetch(kbase + (size_t)(n0 + 256 + ncol) * EMBD, 0, 1);
    }
    v8f c = {0.f, 0.f, 0.f, 0.f, 0.f, 0.f, 0.f, 0.f};
    c = wmma16(aq0, bk0, c);
    c = wmma16(aq1, bk1, c);
#pragma unroll
    for (int j = 0; j < 8; ++j) {
      int r = rsel * 8 + j;
      float val = c[j] * 0.125f;                 // 1/sqrt(64)
      if (m0 + r == n0 + ncol) val = NEG_INF;    // mask self-attention
      S[r * 2048 + n0 + ncol] = val;
    }
    bk0 = nb0;
    bk1 = nb1;
  }
  __syncthreads();

  //---- phase 2: exact softmax over 2048 cols -------------------
  int row = t >> 4, sub = t & 15;
  float mx = NEG_INF;
  for (int c = sub; c < SEQ; c += 16) mx = fmaxf(mx, S[row * 2048 + c]);
  red[row * 16 + sub] = mx;
  __syncthreads();
  if (sub == 0) {
    float m = NEG_INF;
#pragma unroll
    for (int i = 0; i < 16; ++i) m = fmaxf(m, red[row * 16 + i]);
    rowstat[16 + row] = m;
  }
  __syncthreads();
  float rm = rowstat[16 + row];
  float sum = 0.0f;
  for (int c = sub; c < SEQ; c += 16) {
    float p = __expf(S[row * 2048 + c] - rm);
    S[row * 2048 + c] = p;                       // unnormalized P kept in LDS
    sum += p;
  }
  red[row * 16 + sub] = sum;
  __syncthreads();
  if (sub == 0) {
    float s = 0.0f;
#pragma unroll
    for (int i = 0; i < 16; ++i) s += red[row * 16 + i];
    rowstat[row] = 1.0f / s;
  }
  __syncthreads();

  // coalesced normalized attn write (the 537 MB stream)
  float* attn_base = attn + ((size_t)(b * HEADS + h) * SEQ + m0) * SEQ;
  for (int idx = t; idx < 16 * SEQ; idx += 256) {
    int r = idx >> 11, c = idx & 2047;
    attn_base[(size_t)r * SEQ + c] = S[r * 2048 + c] * rowstat[r];
  }
  __syncthreads();

  //---- phase 3: ctx = (P * V) * inv_sum, WMMA over K=2048 ------
  int tile = wv & 3;        // which 16 cols of HEAD
  int khalf = wv >> 2;      // split K range across wave pairs
  const _Float16* vtb =
      Vt + ((size_t)(b * HEADS + h) * HEAD + tile * 16 + ncol) * SEQ + hsel * 16;
  v8f c = {0.f, 0.f, 0.f, 0.f, 0.f, 0.f, 0.f, 0.f};
  v16h bf = *(const v16h*)(vtb + khalf * 32 * 32);
#pragma unroll
  for (int s = 0; s < 32; ++s) {
    int n0 = (khalf * 32 + s) * 32;
    v16h bf_nxt = bf;
    if (s < 31) bf_nxt = *(const v16h*)(vtb + n0 + 32);
    v16h a;
#pragma unroll
    for (int i = 0; i < 8; ++i) {
      a[i]     = (_Float16)S[mrow * 2048 + n0 + hsel * 16 + i];
      a[8 + i] = (_Float16)S[mrow * 2048 + n0 + hsel * 16 + 8 + i];
    }
    c = wmma16(a, bf, c);
    bf = bf_nxt;
  }
  if (khalf == 1) {
#pragma unroll
    for (int j = 0; j < 8; ++j) pvred[(tile * 32 + l) * 8 + j] = c[j];
  }
  __syncthreads();
  if (khalf == 0) {
#pragma unroll
    for (int j = 0; j < 8; ++j) {
      int r = rsel * 8 + j;
      float v = (c[j] + pvred[(tile * 32 + l) * 8 + j]) * rowstat[r];
      ctxh[(size_t)(b * SEQ + m0 + r) * EMBD + h * HEAD + tile * 16 + ncol] =
          (_Float16)v;
    }
  }
}

//==============================================================
// 4) output projection + bias + residual (f32 out), pipelined
//==============================================================
__global__ void __launch_bounds__(256) outproj_kernel(
    const _Float16* __restrict__ ctxh, const _Float16* __restrict__ woh,
    const float* __restrict__ bo, const float* __restrict__ x,
    float* __restrict__ outp) {
  int t = threadIdx.x;
  int l = t & 31;
  int w = blockIdx.x * 8 + (t >> 5);   // 4096 waves
  int mt = w >> 3, es = w & 7;
  int m0 = mt * 16, e0 = es * 64;
  int mrow = l & 15, hsel = l >> 4, ncol = l & 15, rsel = l >> 4;

  const _Float16* arow = ctxh + (size_t)(m0 + mrow) * EMBD;
  const _Float16* wrow = woh + (size_t)(e0 + ncol) * EMBD + hsel * 16;
  v8f acc[4];
#pragma unroll
  for (int n = 0; n < 4; ++n)
#pragma unroll
    for (int j = 0; j < 8; ++j) acc[n][j] = 0.0f;

  v16h a_cur = load_a_frag(arow, hsel);
  v16h b_cur[4];
#pragma unroll
  for (int n = 0; n < 4; ++n)
    b_cur[n] = *(const v16h*)(wrow + (size_t)n * 16 * EMBD);

#pragma unroll
  for (int ks = 0; ks < EMBD / 32; ++ks) {
    v16h a_nxt = a_cur;
    v16h b_nxt[4];
#pragma unroll
    for (int n = 0; n < 4; ++n) b_nxt[n] = b_cur[n];
    if (ks < EMBD / 32 - 1) {
      int k1 = (ks + 1) * 32;
      a_nxt = load_a_frag(arow + k1, hsel);
#pragma unroll
      for (int n = 0; n < 4; ++n)
        b_nxt[n] = *(const v16h*)(wrow + (size_t)n * 16 * EMBD + k1);
    }
#pragma unroll
    for (int n = 0; n < 4; ++n) acc[n] = wmma16(a_cur, b_cur[n], acc[n]);
    a_cur = a_nxt;
#pragma unroll
    for (int n = 0; n < 4; ++n) b_cur[n] = b_nxt[n];
  }

#pragma unroll
  for (int n = 0; n < 4; ++n) {
    int ge = e0 + n * 16 + ncol;
    float bb = bo[ge];
#pragma unroll
    for (int j = 0; j < 8; ++j) {
      size_t idx = (size_t)(m0 + rsel * 8 + j) * EMBD + ge;
      outp[idx] = acc[n][j] + bb + x[idx];
    }
  }
}

//==============================================================
// launch
//==============================================================
extern "C" void kernel_launch(void* const* d_in, const int* in_sizes, int n_in,
                              void* d_out, int out_size, void* d_ws,
                              size_t ws_size, hipStream_t stream) {
  (void)in_sizes; (void)n_in; (void)out_size; (void)ws_size;
  const float* x     = (const float*)d_in[0];
  const float* wq    = (const float*)d_in[1];
  const float* bq    = (const float*)d_in[2];
  const float* wk    = (const float*)d_in[3];
  const float* bk    = (const float*)d_in[4];
  const float* wv    = (const float*)d_in[5];
  const float* bv    = (const float*)d_in[6];
  const float* wo    = (const float*)d_in[7];
  const float* bo    = (const float*)d_in[8];
  const float* gamma = (const float*)d_in[9];
  const float* beta  = (const float*)d_in[10];

  char* ws = (char*)d_ws;
  size_t off = 0;
  _Float16* normh = (_Float16*)(ws + off); off += (size_t)ROWS * EMBD * 2;
  _Float16* Qh    = (_Float16*)(ws + off); off += (size_t)ROWS * EMBD * 2;
  _Float16* Kh    = (_Float16*)(ws + off); off += (size_t)ROWS * EMBD * 2;
  _Float16* Vt    = (_Float16*)(ws + off); off += (size_t)ROWS * EMBD * 2;
  _Float16* ctxh  = (_Float16*)(ws + off); off += (size_t)ROWS * EMBD * 2;
  _Float16* wqh   = (_Float16*)(ws + off); off += (size_t)EMBD * EMBD * 2;
  _Float16* wkh   = (_Float16*)(ws + off); off += (size_t)EMBD * EMBD * 2;
  _Float16* wvh   = (_Float16*)(ws + off); off += (size_t)EMBD * EMBD * 2;
  _Float16* woh   = (_Float16*)(ws + off); off += (size_t)EMBD * EMBD * 2;

  float* outp = (float*)d_out;
  float* attn = outp + (size_t)ROWS * EMBD;

  cvt_w_kernel<<<(EMBD * EMBD + 255) / 256, 256, 0, stream>>>(
      wq, wk, wv, wo, wqh, wkh, wvh, woh);
  ln_kernel<<<ROWS, 256, 0, stream>>>(x, gamma, beta, normh);
  proj_qkv_kernel<<<3 * 512 * 8 / 8, 256, 0, stream>>>(
      normh, wqh, wkh, wvh, bq, bk, bv, Qh, Kh, Vt);
  attn_kernel<<<BATCH * HEADS * (SEQ / 16), 256,
                SMEM_FLOATS * sizeof(float), stream>>>(Qh, Kh, Vt, attn, ctxh);
  outproj_kernel<<<512 * 8 / 8, 256, 0, stream>>>(ctxh, woh, bo, x, outp);
}
